// EncLayer_1194000908868
// MI455X (gfx1250) — compile-verified, matrix operand order checked by reading, not run on hardware
//
#include <hip/hip_runtime.h>
#include <hip/hip_bf16.h>
#include <math.h>
#include <stdint.h>

typedef __bf16 bf16_t;
typedef __attribute__((ext_vector_type(16))) __bf16 v16bf;
typedef __attribute__((ext_vector_type(8)))  __bf16 v8bf;
typedef __attribute__((ext_vector_type(8)))  float  v8f;
typedef __attribute__((ext_vector_type(4)))  unsigned int u32x4;
typedef __attribute__((ext_vector_type(8)))  int i32x8;
typedef __attribute__((ext_vector_type(4)))  int i32x4;

#define B_   32
#define L_   512
#define D_   128
#define H_   8
#define FF1_ 128
#define FF2_ 64
#define ROWS_ (B_*L_)            // 16384 token rows

// LDS B-panel: 64 rows x 128 bf16 per K-chunk, padded rows (272B) -> conflict-free
#define KCHUNK 128
#define PANEL_ROWS 64
#define PANEL_STRIDE 136          // elements; 272B = 68 banks
#define PANEL_ELEMS (PANEL_ROWS*PANEL_STRIDE)

#if __has_builtin(__builtin_amdgcn_tensor_load_to_lds) && __has_builtin(__builtin_amdgcn_s_wait_tensorcnt)
#define USE_TDM 1
#else
#define USE_TDM 0
#endif

union AFrag { v16bf v; v8bf h[2]; };

// Stage one 64 x 128 bf16 panel chunk from global (row stride ldb elems) into LDS.
// TDM path: one wave issues a 2D tensor DMA with LDS padding (256B data + 16B pad
// per row -> PANEL_STRIDE). Fallback: cooperative load + ds_store.
__device__ __forceinline__ void stage_panel(const bf16_t* __restrict__ gsrc, int ldb,
                                            bf16_t* panel, int buf)
{
  bf16_t* dst = panel + buf * PANEL_ELEMS;
#if USE_TDM
  if ((threadIdx.x >> 5) == 0) {
    unsigned long long ga = (unsigned long long)(uintptr_t)gsrc;
    unsigned ldsoff = (unsigned)(uintptr_t)dst;   // flat LDS addr truncates to LDS offset
    u32x4 g0 = { 1u,                               // count=1, gather off
                 ldsoff,                           // D#.lds_addr
                 (unsigned)ga,                     // D#.global_addr[31:0]
                 (unsigned)((ga >> 32) & 0x01FFFFFFu) | 0x80000000u }; // addr[56:32] | type=2
    // word0: data_size=1(2B) | pad_enable | pad_interval=5(64 dw) | pad_amount=3(4 dw)
    int w0 = (1 << 16) | (1 << 20) | (5 << 22) | (3 << 25);
    unsigned dim0 = 0x40000000u, dim1 = 0x40000000u;  // huge tensor dims (tile in-bounds)
    i32x8 g1 = { w0,
                 (int)((dim0 & 0xFFFFu) << 16),                        // dim0[15:0]
                 (int)((dim0 >> 16) | ((dim1 & 0xFFFFu) << 16)),       // dim0[31:16], dim1[15:0]
                 (int)((dim1 >> 16) | (128u << 16)),                   // dim1[31:16], tile_dim0=128
                 (int)(64u),                                           // tile_dim1=64, tile_dim2=0
                 (int)(unsigned)ldb,                                   // dim0_stride[31:0]
                 0, 0 };
    i32x4 gz4 = { 0, 0, 0, 0 };
    i32x8 gz8 = { 0, 0, 0, 0, 0, 0, 0, 0 };
    __builtin_amdgcn_tensor_load_to_lds(g0, g1, gz4, gz4, gz8, 0);
  }
#else
  const int tid = threadIdx.x;
#pragma unroll
  for (int j = 0; j < 4; ++j) {
    int i = tid + 256 * j;                 // 1024 x 16B transfers
    int row = i >> 4, e = (i & 15) * 8;
    *(v8bf*)(dst + row * PANEL_STRIDE + e) = *(const v8bf*)(gsrc + (long)row * ldb + e);
  }
#endif
}

__device__ __forceinline__ void stage_wait(int pending) {
#if USE_TDM
  if ((threadIdx.x >> 5) == 0) {
    if (pending) __builtin_amdgcn_s_wait_tensorcnt(1);
    else         __builtin_amdgcn_s_wait_tensorcnt(0);
  }
#else
  (void)pending;
#endif
}

// Block(256 thr, 8 waves) GEMM: each wave computes a 32x64 tile of
// C = A(MxK,row-major) * Bt(NxK,row-major)^T. B panel staged via LDS (TDM),
// double buffered across K chunks of 128. K must be a multiple of 128.
__device__ __forceinline__ void block_gemm_32x64(
    const bf16_t* __restrict__ A,   // offset to m0*lda (wave's 32 rows)
    const bf16_t* __restrict__ Bt,  // offset to n0*ldb (block's 64 rows)
    int lda, int ldb, int K, bf16_t* panel, v8f acc[2][4])
{
  const int lane = threadIdx.x & 31;
  const int r  = lane & 15;
  const int kb = (lane >> 4) << 3;                 // 0 or 8
  const bf16_t* ap0 = A + (long)r * lda + kb;
  const bf16_t* ap1 = A + (long)(r + 16) * lda + kb;
  const int nch = K / KCHUNK;

  stage_panel(Bt, ldb, panel, 0);
  for (int c = 0; c < nch; ++c) {
    if (c + 1 < nch) stage_panel(Bt + (long)(c + 1) * KCHUNK, ldb, panel, (c + 1) & 1);
    stage_wait((c + 1 < nch) ? 1 : 0);
    __syncthreads();
    const bf16_t* pb = panel + (c & 1) * PANEL_ELEMS;
    const long kg = (long)c * KCHUNK;
#pragma unroll
    for (int ks = 0; ks < 4; ++ks) {
      const int k0 = ks * 32;
      AFrag a0, a1;
      a0.h[0] = *(const v8bf*)(ap0 + kg + k0);
      a0.h[1] = *(const v8bf*)(ap0 + kg + k0 + 16);
      a1.h[0] = *(const v8bf*)(ap1 + kg + k0);
      a1.h[1] = *(const v8bf*)(ap1 + kg + k0 + 16);
#pragma unroll
      for (int t = 0; t < 4; ++t) {
        AFrag b;
        const bf16_t* lp = pb + (t * 16 + r) * PANEL_STRIDE + kb + k0;
        b.h[0] = *(const v8bf*)(lp);
        b.h[1] = *(const v8bf*)(lp + 16);
        acc[0][t] = __builtin_amdgcn_wmma_f32_16x16x32_bf16(
            false, a0.v, false, b.v, (short)0, acc[0][t], false, false);
        acc[1][t] = __builtin_amdgcn_wmma_f32_16x16x32_bf16(
            false, a1.v, false, b.v, (short)0, acc[1][t], false, false);
      }
    }
    __syncthreads();
  }
}

// Direct (no-LDS) 16x64 wave GEMM for small K (ffn3, K=64)
__device__ __forceinline__ void wave_gemm_16x64(
    const bf16_t* __restrict__ A, const bf16_t* __restrict__ Bt,
    int lda, int ldb, int K, v8f acc[4])
{
  const int lane = threadIdx.x & 31;
  const int r  = lane & 15;
  const int kb = (lane >> 4) << 3;
  const bf16_t* ap  = A  + (long)r * lda + kb;
  const bf16_t* bp0 = Bt + (long)r * ldb + kb;
  for (int k0 = 0; k0 < K; k0 += 32) {
    AFrag a;
    a.h[0] = *(const v8bf*)(ap + k0);
    a.h[1] = *(const v8bf*)(ap + k0 + 16);
#pragma unroll
    for (int t = 0; t < 4; ++t) {
      AFrag b;
      const bf16_t* bp = bp0 + (long)t * 16 * ldb + k0;
      b.h[0] = *(const v8bf*)(bp);
      b.h[1] = *(const v8bf*)(bp + 16);
      acc[t] = __builtin_amdgcn_wmma_f32_16x16x32_bf16(
          false, a.v, false, b.v, (short)0, acc[t], false, false);
    }
  }
}

// ---------------- prep kernels ----------------
__global__ void k_f32_to_bf16(const float* __restrict__ in, bf16_t* __restrict__ out, int n) {
  int i = blockIdx.x * blockDim.x + threadIdx.x;
  if (i < n) out[i] = (bf16_t)in[i];
}

__global__ void k_transpose_bf16(const float* __restrict__ in, bf16_t* __restrict__ out,
                                 int K, int N) {
  long base = (long)blockIdx.z * K * N;
  int o = blockIdx.x * blockDim.x + threadIdx.x;
  if (o < K * N) {
    int n = o / K, k = o - n * K;
    out[base + o] = (bf16_t)in[base + (long)k * N + n];
  }
}

// ---------------- QKV projection: z = mat*H + h --------------------
__global__ void k_qkv(const bf16_t* __restrict__ xbf, const bf16_t* __restrict__ wqkv_t,
                      const float* __restrict__ bq, const float* __restrict__ bk,
                      const float* __restrict__ bv,
                      bf16_t* __restrict__ q, bf16_t* __restrict__ k, bf16_t* __restrict__ vt)
{
  __shared__ bf16_t panel[2 * PANEL_ELEMS];
  int z = blockIdx.z;
  int mat = z / H_, h = z - mat * H_;
  int wave = threadIdx.x >> 5, lane = threadIdx.x & 31;
  int m0 = blockIdx.y * 256 + wave * 32;
  int n0 = blockIdx.x * 64;
  v8f acc[2][4] = {};
  block_gemm_32x64(xbf + (long)m0 * D_, wqkv_t + (long)z * D_ * D_ + (long)n0 * D_,
                   D_, D_, D_, panel, acc);
  const float* bias = (mat == 0) ? bq : (mat == 1) ? bk : bv;
  bias += h * D_;
  int col = lane & 15, rowoff = (lane >> 4) * 8;
#pragma unroll
  for (int mi = 0; mi < 2; ++mi)
#pragma unroll
    for (int t = 0; t < 4; ++t)
#pragma unroll
      for (int r = 0; r < 8; ++r) {
        int row = m0 + mi * 16 + rowoff + r;
        int b = row >> 9, l = row & (L_ - 1);
        int n = n0 + t * 16 + col;
        float v = acc[mi][t][r] + bias[n];
        long bh = (long)(b * H_ + h);
        if (mat == 0)      q[(bh * L_ + l) * D_ + n] = (bf16_t)v;
        else if (mat == 1) k[(bh * L_ + l) * D_ + n] = (bf16_t)v;
        else               vt[(bh * D_ + n) * L_ + l] = (bf16_t)v;
      }
}

// ---------------- scores = mask(Q K^T)/sqrt(D) ----------------
__global__ void k_scores(const bf16_t* __restrict__ q, const bf16_t* __restrict__ kk,
                         const int* __restrict__ mask, bf16_t* __restrict__ sc)
{
  __shared__ bf16_t panel[2 * PANEL_ELEMS];
  int z = blockIdx.z, b = z >> 3;
  int wave = threadIdx.x >> 5, lane = threadIdx.x & 31;
  int m0 = blockIdx.y * 256 + wave * 32;
  int n0 = blockIdx.x * 64;
  v8f acc[2][4] = {};
  block_gemm_32x64(q + (long)z * L_ * D_ + (long)m0 * D_,
                   kk + (long)z * L_ * D_ + (long)n0 * D_, D_, D_, D_, panel, acc);
  const float scale = 0.08838834764831845f;   // 1/sqrt(128)
  int col = lane & 15, rowoff = (lane >> 4) * 8;
  long scb = (long)z * L_ * L_, mb = (long)b * L_ * L_;
#pragma unroll
  for (int mi = 0; mi < 2; ++mi)
#pragma unroll
    for (int t = 0; t < 4; ++t)
#pragma unroll
      for (int r = 0; r < 8; ++r) {
        int qr = m0 + mi * 16 + rowoff + r;
        int kc = n0 + t * 16 + col;
        float s = acc[mi][t][r];
        if (mask[mb + (long)qr * L_ + kc] != 0) s = -1e9f;   // mask BEFORE scale
        s *= scale;
        sc[scb + (long)qr * L_ + kc] = (bf16_t)s;
      }
}

// ---------------- per-column (query-axis) softmax stats ----------------
__global__ void k_colstats(const bf16_t* __restrict__ sc,
                           float* __restrict__ cmax, float* __restrict__ csum) {
  int g = blockIdx.x * blockDim.x + threadIdx.x;
  if (g >= B_ * H_ * L_) return;
  int z = g >> 9, kc = g & (L_ - 1);
  const bf16_t* p = sc + (long)z * L_ * L_ + kc;
  float m = -3.0e38f, s = 0.f;
  for (int qi = 0; qi < L_; ++qi) {
    float v = (float)p[(long)qi * L_];
    float mn = fmaxf(m, v);
    s = s * __expf(m - mn) + __expf(v - mn);
    m = mn;
  }
  cmax[g] = m; csum[g] = s;
}

__global__ void k_softmax_norm(bf16_t* __restrict__ sc, const float* __restrict__ cmax,
                               const float* __restrict__ csum) {
  long i = (long)blockIdx.x * blockDim.x + threadIdx.x;
  if (i >= (long)B_ * H_ * L_ * L_) return;
  long z = i >> 18;
  int kc = (int)(i & (L_ - 1));
  int st = (int)z * L_ + kc;
  float v = __expf((float)sc[i] - cmax[st]) / csum[st];
  sc[i] = (bf16_t)v;
}

// ---------------- att = attn @ V (stored [b][l][h][d]) -----------
__global__ void k_attv(const bf16_t* __restrict__ sc, const bf16_t* __restrict__ vt,
                       bf16_t* __restrict__ attcat) {
  __shared__ bf16_t panel[2 * PANEL_ELEMS];
  int z = blockIdx.z, b = z >> 3, h = z & 7;
  int wave = threadIdx.x >> 5, lane = threadIdx.x & 31;
  int m0 = blockIdx.y * 256 + wave * 32;
  int n0 = blockIdx.x * 64;
  v8f acc[2][4] = {};
  block_gemm_32x64(sc + (long)z * L_ * L_ + (long)m0 * L_,
                   vt + (long)z * D_ * L_ + (long)n0 * L_, L_, L_, L_, panel, acc);
  int col = lane & 15, rowoff = (lane >> 4) * 8;
#pragma unroll
  for (int mi = 0; mi < 2; ++mi)
#pragma unroll
    for (int t = 0; t < 4; ++t)
#pragma unroll
      for (int r = 0; r < 8; ++r) {
        int qr = m0 + mi * 16 + rowoff + r;
        int d  = n0 + t * 16 + col;
        attcat[(long)(b * L_ + qr) * (H_ * D_) + h * D_ + d] = (bf16_t)acc[mi][t][r];
      }
}

// ---------------- generic staged GEMM (K multiple of 128) ---------
__global__ void k_gemm(const bf16_t* __restrict__ A, const bf16_t* __restrict__ Bt,
                       const float* __restrict__ bias, int K,
                       float* __restrict__ Cf, bf16_t* __restrict__ Cb, int ldc, int relu)
{
  __shared__ bf16_t panel[2 * PANEL_ELEMS];
  int wave = threadIdx.x >> 5, lane = threadIdx.x & 31;
  int m0 = blockIdx.y * 256 + wave * 32;
  int n0 = blockIdx.x * 64;
  v8f acc[2][4] = {};
  block_gemm_32x64(A + (long)m0 * K, Bt + (long)n0 * K, K, K, K, panel, acc);
  int col = lane & 15, rowoff = (lane >> 4) * 8;
#pragma unroll
  for (int mi = 0; mi < 2; ++mi)
#pragma unroll
    for (int t = 0; t < 4; ++t)
#pragma unroll
      for (int r = 0; r < 8; ++r) {
        int row = m0 + mi * 16 + rowoff + r, n = n0 + t * 16 + col;
        float v = acc[mi][t][r] + bias[n];
        if (relu) v = fmaxf(v, 0.f);
        if (Cf) Cf[(long)row * ldc + n] = v;
        if (Cb) Cb[(long)row * ldc + n] = (bf16_t)v;
      }
}

// small-K GEMM (ffn3, K=64), direct loads
__global__ void k_gemm_small(const bf16_t* __restrict__ A, const bf16_t* __restrict__ Bt,
                             const float* __restrict__ bias, int K,
                             float* __restrict__ Cf, int ldc)
{
  int wave = threadIdx.x >> 5, lane = threadIdx.x & 31;
  int m0 = blockIdx.y * 128 + wave * 16;
  int n0 = blockIdx.x * 64;
  v8f acc[4] = {};
  wave_gemm_16x64(A + (long)m0 * K, Bt + (long)n0 * K, K, K, K, acc);
  int col = lane & 15, rowoff = (lane >> 4) * 8;
#pragma unroll
  for (int t = 0; t < 4; ++t)
#pragma unroll
    for (int r = 0; r < 8; ++r) {
      int row = m0 + rowoff + r, n = n0 + t * 16 + col;
      Cf[(long)row * ldc + n] = acc[t][r] + bias[n];
    }
}

// ---------------- layernorm (+residual) ----------------
__global__ void k_ln(const float* __restrict__ y, const float* __restrict__ addv,
                     const float* __restrict__ g, const float* __restrict__ bb,
                     float* __restrict__ fout, bf16_t* __restrict__ bfout)
{
  int row  = blockIdx.x * 8 + (threadIdx.x >> 5);
  int lane = threadIdx.x & 31;
  float4 yv = ((const float4*)(y + (long)row * D_))[lane];
  float s = yv.x + yv.y + yv.z + yv.w;
  float q = yv.x*yv.x + yv.y*yv.y + yv.z*yv.z + yv.w*yv.w;
  for (int m = 16; m >= 1; m >>= 1) { s += __shfl_xor(s, m, 32); q += __shfl_xor(q, m, 32); }
  float mean = s * (1.f / D_);
  float var  = q * (1.f / D_) - mean * mean;
  float inv  = rsqrtf(var + 1e-6f);
  float4 av = ((const float4*)(addv + (long)row * D_))[lane];
  float yy[4] = {yv.x, yv.y, yv.z, yv.w};
  float aa[4] = {av.x, av.y, av.z, av.w};
  float ov[4], lv[4];
#pragma unroll
  for (int c = 0; c < 4; ++c) {
    int colc = lane * 4 + c;
    float ln = (yy[c] - mean) * inv * g[colc] + bb[colc];
    lv[c] = ln; ov[c] = ln + aa[c];
  }
  ((float4*)(fout + (long)row * D_))[lane] = make_float4(ov[0], ov[1], ov[2], ov[3]);
  if (bfout) {
#pragma unroll
    for (int c = 0; c < 4; ++c)
      bfout[(long)row * D_ + lane * 4 + c] = (bf16_t)lv[c];
  }
}

extern "C" void kernel_launch(void* const* d_in, const int* in_sizes, int n_in,
                              void* d_out, int out_size, void* d_ws, size_t ws_size,
                              hipStream_t stream) {
  (void)in_sizes; (void)n_in; (void)out_size; (void)ws_size;
  const float* x    = (const float*)d_in[0];
  const int*   mask = (const int*)  d_in[1];
  const float* Wq = (const float*)d_in[4];  const float* bq = (const float*)d_in[5];
  const float* Wk = (const float*)d_in[6];  const float* bk = (const float*)d_in[7];
  const float* Wv = (const float*)d_in[8];  const float* bv = (const float*)d_in[9];
  const float* Wo = (const float*)d_in[10]; const float* bo = (const float*)d_in[11];
  const float* W1 = (const float*)d_in[12]; const float* b1 = (const float*)d_in[13];
  const float* W2 = (const float*)d_in[14]; const float* b2 = (const float*)d_in[15];
  const float* W3 = (const float*)d_in[16]; const float* b3 = (const float*)d_in[17];
  const float* lg = (const float*)d_in[18]; const float* lb = (const float*)d_in[19];

  char* p = (char*)d_ws;
  auto alloc = [&](size_t bytes) { void* r = (void*)p; p += (bytes + 255) & ~(size_t)255; return r; };
  bf16_t* xbf    = (bf16_t*)alloc((size_t)ROWS_ * D_ * 2);
  bf16_t* wqkv_t = (bf16_t*)alloc((size_t)3 * H_ * D_ * D_ * 2);
  bf16_t* wo_t   = (bf16_t*)alloc((size_t)D_ * H_ * D_ * 2);
  bf16_t* w1_t   = (bf16_t*)alloc((size_t)FF1_ * D_ * 2);
  bf16_t* w2_t   = (bf16_t*)alloc((size_t)FF2_ * FF1_ * 2);
  bf16_t* w3_t   = (bf16_t*)alloc((size_t)D_ * FF2_ * 2);
  bf16_t* qbf    = (bf16_t*)alloc((size_t)B_ * H_ * L_ * D_ * 2);
  bf16_t* kbf    = (bf16_t*)alloc((size_t)B_ * H_ * L_ * D_ * 2);
  bf16_t* vtbf   = (bf16_t*)alloc((size_t)B_ * H_ * L_ * D_ * 2);
  bf16_t* sc     = (bf16_t*)alloc((size_t)B_ * H_ * L_ * L_ * 2);
  float*  cmax   = (float*) alloc((size_t)B_ * H_ * L_ * 4);
  float*  csum   = (float*) alloc((size_t)B_ * H_ * L_ * 4);
  bf16_t* attcat = (bf16_t*)alloc((size_t)ROWS_ * H_ * D_ * 2);
  float*  attlin = (float*) alloc((size_t)ROWS_ * D_ * 4);
  float*  res2   = (float*) alloc((size_t)ROWS_ * D_ * 4);
  bf16_t* x1bf   = (bf16_t*)alloc((size_t)ROWS_ * D_ * 2);
  bf16_t* h1bf   = (bf16_t*)alloc((size_t)ROWS_ * FF1_ * 2);
  bf16_t* h2bf   = (bf16_t*)alloc((size_t)ROWS_ * FF2_ * 2);
  float*  ff     = (float*) alloc((size_t)ROWS_ * D_ * 4);

  // prep: x -> bf16; weights -> transposed bf16 [N][K]
  k_f32_to_bf16<<<(ROWS_ * D_ + 255) / 256, 256, 0, stream>>>(x, xbf, ROWS_ * D_);
  k_transpose_bf16<<<dim3((D_*D_ + 255)/256, 1, H_), 256, 0, stream>>>(Wq, wqkv_t, D_, D_);
  k_transpose_bf16<<<dim3((D_*D_ + 255)/256, 1, H_), 256, 0, stream>>>(Wk, wqkv_t + (size_t)H_*D_*D_, D_, D_);
  k_transpose_bf16<<<dim3((D_*D_ + 255)/256, 1, H_), 256, 0, stream>>>(Wv, wqkv_t + (size_t)2*H_*D_*D_, D_, D_);
  k_transpose_bf16<<<dim3((H_*D_*D_ + 255)/256, 1, 1), 256, 0, stream>>>(Wo, wo_t, H_*D_, D_);
  k_transpose_bf16<<<dim3((D_*FF1_ + 255)/256, 1, 1), 256, 0, stream>>>(W1, w1_t, D_, FF1_);
  k_transpose_bf16<<<dim3((FF1_*FF2_ + 255)/256, 1, 1), 256, 0, stream>>>(W2, w2_t, FF1_, FF2_);
  k_transpose_bf16<<<dim3((FF2_*D_ + 255)/256, 1, 1), 256, 0, stream>>>(W3, w3_t, FF2_, D_);

  // attention pipeline
  k_qkv<<<dim3(2, ROWS_/256, 3*H_), 256, 0, stream>>>(xbf, wqkv_t, bq, bk, bv, qbf, kbf, vtbf);
  k_scores<<<dim3(L_/64, L_/256, B_*H_), 256, 0, stream>>>(qbf, kbf, mask, sc);
  k_colstats<<<dim3(B_*H_*L_/256), 256, 0, stream>>>(sc, cmax, csum);
  {
    long total = (long)B_ * H_ * L_ * L_;
    k_softmax_norm<<<dim3((unsigned)((total + 255) / 256)), 256, 0, stream>>>(sc, cmax, csum);
  }
  k_attv<<<dim3(D_/64, L_/256, B_*H_), 256, 0, stream>>>(sc, vtbf, attcat);
  k_gemm<<<dim3(2, ROWS_/256), 256, 0, stream>>>(attcat, wo_t, bo, H_*D_, attlin, nullptr, D_, 0);

  // LN1 + residual, FFN, LN2 + residual -> d_out (fp32)
  k_ln<<<dim3(ROWS_/8), 256, 0, stream>>>(attlin, x, lg, lb, res2, x1bf);
  k_gemm<<<dim3(2, ROWS_/256), 256, 0, stream>>>(x1bf, w1_t, b1, D_,   nullptr, h1bf, FF1_, 1);
  k_gemm<<<dim3(1, ROWS_/256), 256, 0, stream>>>(h1bf, w2_t, b2, FF1_, nullptr, h2bf, FF2_, 1);
  k_gemm_small<<<dim3(2, ROWS_/128), 256, 0, stream>>>(h2bf, w3_t, b3, FF2_, ff, D_);
  k_ln<<<dim3(ROWS_/8), 256, 0, stream>>>(ff, res2, lg, lb, (float*)d_out, nullptr);
}
